// RNN_7078106104561
// MI455X (gfx1250) — matrix operand order repaired and checked by reading, not compiled, not run on hardware
//
#include <hip/hip_runtime.h>

// ---------------------------------------------------------------------------
// 2-layer LSTM LM for MI455X (gfx1250, wave32, WMMA, async-LDS staging).
// decoded[s,b,v], h1[b,h], h2[b,h] concatenated in d_out.
// ---------------------------------------------------------------------------

typedef __attribute__((ext_vector_type(16))) __bf16 v16bf;
typedef __attribute__((ext_vector_type(8)))  float  v8f;

#define SEQ   256
#define BATCH 32
#define SB    (SEQ*BATCH)     // 8192 rows
#define NIN   256
#define NH    256
#define NHG   (NH*4)          // 1024 gate columns (n = g*256 + h)
#define NVOC  10000
#define MT_TOT (SB/16)        // 512 M-tiles
#define NT_G   (NHG/16)       // 64 gate N-tiles
#define KC     (NH/32)        // 8 K-chunks of 32
#define NT_V   (NVOC/16)      // 625 vocab N-tiles (exact)

#define OUT_DEC  0
#define OUT_H1   (SB*(size_t)NVOC)          // 81,920,000
#define OUT_H2   (OUT_H1 + BATCH*NH)        // +8192

static __device__ __forceinline__ v8f splat8(float x) {
  v8f r;
  #pragma unroll
  for (int i = 0; i < 8; ++i) r[i] = x;
  return r;
}

// A-operand (16-bit, 16x32 MxK) per-lane element -> K index inside 32-chunk.
static __device__ __forceinline__ int a_elem_k(int e, int half) {
  int i = e >> 1, lsb = e & 1;
  return ((i < 4) ? 2 * i : 16 + 2 * (i - 4)) + 8 * half + lsb;
}
// Inverse: K-in-chunk -> (half, elem)
static __device__ __forceinline__ void a_inv(int kl, int& half, int& e) {
  int q = kl >> 3;
  half = q & 1;
  int i = ((kl & 7) >> 1) + ((q >= 2) ? 4 : 0);
  e = 2 * i + (kl & 1);
}

static __device__ __forceinline__ v8f wmma_bf16(v16bf a, v16bf b, v8f c) {
  return __builtin_amdgcn_wmma_f32_16x16x32_bf16(false, a, false, b,
                                                 (short)0, c, false, false);
}

// ---- gfx1250 async global->LDS copy (ASYNCcnt-tracked, bypasses VGPRs) -----
// GVS mode: lds_byte_off (VGPR), voffset (VGPR), saddr (SGPR pair).
static __device__ __forceinline__ void async_copy_b128(unsigned lds_off,
                                                       const void* sbase,
                                                       unsigned voff) {
  asm volatile("global_load_async_to_lds_b128 %0, %1, %2"
               :: "v"(lds_off), "v"(voff), "s"(sbase)
               : "memory");
}
static __device__ __forceinline__ void wait_async0() {
  asm volatile("s_wait_asynccnt 0x0" ::: "memory");
}

// ---------------- weight packing -------------------------------------------
// Gate weights W[k][h][g] -> bf16 B tiles, n = g*256+h.
// B layout: n=lane%16; elem e<8: K=8*half+e, e>=8: K=16+8*half+(e-8).
__global__ void k_pack_gateW(const float* __restrict__ W, __bf16* __restrict__ out) {
  int tid = blockIdx.x * blockDim.x + threadIdx.x;   // 8*64*32 = 16384
  if (tid >= KC * NT_G * 32) return;
  int lane = tid & 31, nt = (tid >> 5) & (NT_G - 1), kc = tid >> 11;
  int half = lane >> 4;
  int n = nt * 16 + (lane & 15);
  int h = n & (NH - 1), g = n >> 8;
  __bf16* o = out + (size_t)tid * 16;
  #pragma unroll
  for (int e = 0; e < 16; ++e) {
    int k = kc * 32 + ((e < 8) ? 8 * half + e : 16 + 8 * half + (e - 8));
    o[e] = (__bf16)W[(size_t)k * NHG + h * 4 + g];
  }
}

// dec_W[v][h] -> bf16 B tiles over n=v, k=h.
__global__ void k_pack_decW(const float* __restrict__ W, __bf16* __restrict__ out) {
  int tid = blockIdx.x * blockDim.x + threadIdx.x;   // 8*625*32 = 160000
  if (tid >= KC * NT_V * 32) return;
  int lane = tid & 31;
  int rem = tid >> 5;                 // kc*625 + nt
  int nt = rem % NT_V, kc = rem / NT_V;
  int half = lane >> 4;
  int v = nt * 16 + (lane & 15);
  __bf16* o = out + (size_t)tid * 16;
  #pragma unroll
  for (int e = 0; e < 16; ++e) {
    int k = kc * 32 + ((e < 8) ? 8 * half + e : 16 + 8 * half + (e - 8));
    o[e] = (__bf16)W[(size_t)v * NH + k];
  }
}

// bias[h][4] -> bias_packed[n = g*256+h]
__global__ void k_pack_bias(const float* __restrict__ b, float* __restrict__ out) {
  int n = blockIdx.x * blockDim.x + threadIdx.x;
  if (n < NHG) out[n] = b[(n & (NH - 1)) * 4 + (n >> 8)];
}

// Embedding gather straight into bf16 A-tile layout.
__global__ void k_embed_pack(const int* __restrict__ tokens,
                             const float* __restrict__ embW,
                             __bf16* __restrict__ xp) {
  int tid = blockIdx.x * blockDim.x + threadIdx.x;   // 512*8*32 = 131072
  if (tid >= MT_TOT * KC * 32) return;
  int lane = tid & 31, kc = (tid >> 5) & 7, mt = tid >> 8;
  int half = lane >> 4;
  int sb = mt * 16 + (lane & 15);                    // sb = s*32 + b
  const float* row = embW + (size_t)tokens[sb] * NIN;
  __bf16* o = xp + (size_t)tid * 16;
  #pragma unroll
  for (int e = 0; e < 16; ++e)
    o[e] = (__bf16)row[kc * 32 + a_elem_k(e, half)];
}

// ---------------- gx = X(A-packed bf16) x W(B-packed) + bias ----------------
// A panel (8 KB) async-staged to LDS once per block, reused by 8 waves.
// Output in WMMA C/D register layout: gxp[((mt*64+nt)*32+lane)] as v8f.
__global__ __launch_bounds__(256)
void k_gemm_gx(const __bf16* __restrict__ Ap, const __bf16* __restrict__ Bp,
               const float* __restrict__ bias, float* __restrict__ gxp) {
  __shared__ __align__(32) __bf16 alds[KC * 32 * 16];   // 8 KB
  int tid = threadIdx.x;
  int lane = tid & 31, w = tid >> 5;                    // 8 waves, 8 n-tiles each
  int mt = blockIdx.x;

  // cooperative async stage of this mtile's A panel (256 thr x 32 B)
  const void* apanel = (const void*)(Ap + (size_t)mt * KC * 32 * 16);
  unsigned lbase = (unsigned)(size_t)(&alds[0]);
  async_copy_b128(lbase + tid * 32,      apanel, (unsigned)(tid * 32));
  async_copy_b128(lbase + tid * 32 + 16, apanel, (unsigned)(tid * 32 + 16));
  wait_async0();
  __syncthreads();

  const v16bf* Bv = (const v16bf*)Bp;
  v8f acc[8];
  #pragma unroll
  for (int t = 0; t < 8; ++t)
    acc[t] = splat8(bias[(w * 8 + t) * 16 + (lane & 15)]);
  #pragma unroll
  for (int kc = 0; kc < KC; ++kc) {
    v16bf a = *(const v16bf*)&alds[(kc * 32 + lane) * 16];
    #pragma unroll
    for (int t = 0; t < 8; ++t) {
      v16bf b = Bv[((size_t)kc * NT_G + (w * 8 + t)) * 32 + lane];
      acc[t] = wmma_bf16(a, b, acc[t]);
    }
  }
  v8f* g = (v8f*)gxp;
  #pragma unroll
  for (int t = 0; t < 8; ++t)
    g[((size_t)mt * NT_G + (w * 8 + t)) * 32 + lane] = acc[t];
}

// ---------------- sequential LSTM scan (one WGP, 32 waves) ------------------
// wave w: mt = w&1 (batch rows), j = w>>1 (channels 16j..16j+15); its 4 WMMA
// tiles are n-tiles {j, j+16, j+32, j+48} = gates i,f,o,g of the SAME (b,h)
// set -> gate math is lane-local. c lives in registers, h in LDS (A layout).
__global__ __launch_bounds__(1024)
void k_lstm(const float* __restrict__ gxp, const __bf16* __restrict__ Whp,
            __bf16* __restrict__ hseq, float* __restrict__ hout) {
  __shared__ __align__(32) __bf16 hlds[2 * KC * 32 * 16];   // 16 KB, A layout
  int lane = threadIdx.x & 31, wid = threadIdx.x >> 5;
  int mt = wid & 1, j = wid >> 1;

  unsigned int* hz = (unsigned int*)hlds;
  for (int i = threadIdx.x; i < 2 * KC * 32 * 8; i += 1024) hz[i] = 0u;
  __syncthreads();

  float c[8];
  #pragma unroll
  for (int v = 0; v < 8; ++v) c[v] = 0.f;

  // scatter constants for h_new -> A layout (k = hcol)
  int hcol = j * 16 + (lane & 15);
  int kcp = hcol >> 5, shalf, se;
  a_inv(hcol & 31, shalf, se);
  int mbase = 8 * (lane >> 4);

  const v8f*   gv = (const v8f*)gxp;
  const v16bf* Bv = (const v16bf*)Whp;

  for (int t = 0; t < SEQ; ++t) {
    int mtg = 2 * t + mt;
    v8f acc0 = gv[((size_t)mtg * NT_G + j     ) * 32 + lane];
    v8f acc1 = gv[((size_t)mtg * NT_G + j + 16) * 32 + lane];
    v8f acc2 = gv[((size_t)mtg * NT_G + j + 32) * 32 + lane];
    v8f acc3 = gv[((size_t)mtg * NT_G + j + 48) * 32 + lane];
    #pragma unroll
    for (int kc = 0; kc < KC; ++kc) {
      v16bf a = *(const v16bf*)&hlds[((mt * KC + kc) * 32 + lane) * 16];
      acc0 = wmma_bf16(a, Bv[((size_t)kc * NT_G + j     ) * 32 + lane], acc0);
      acc1 = wmma_bf16(a, Bv[((size_t)kc * NT_G + j + 16) * 32 + lane], acc1);
      acc2 = wmma_bf16(a, Bv[((size_t)kc * NT_G + j + 32) * 32 + lane], acc2);
      acc3 = wmma_bf16(a, Bv[((size_t)kc * NT_G + j + 48) * 32 + lane], acc3);
    }
    float hn[8];
    #pragma unroll
    for (int v = 0; v < 8; ++v) {
      float ig = 1.f / (1.f + __expf(-acc0[v]));
      float fg = 1.f / (1.f + __expf(-acc1[v]));
      float og = 1.f / (1.f + __expf(-acc2[v]));
      float gg = tanhf(acc3[v]);
      c[v] = fg * c[v] + ig * gg;
      hn[v] = og * tanhf(c[v]);
    }
    __syncthreads();            // all waves finished reading hlds[t-1]
    __bf16* gs = hseq + ((size_t)(mtg * KC + kcp) * 32) * 16;
    #pragma unroll
    for (int v = 0; v < 8; ++v) {
      int m = mbase + v;                       // row in M-tile
      int lp = m + 16 * shalf;
      __bf16 hb = (__bf16)hn[v];
      hlds[((mt * KC + kcp) * 32 + lp) * 16 + se] = hb;
      gs[lp * 16 + se] = hb;
    }
    if (t == SEQ - 1) {
      #pragma unroll
      for (int v = 0; v < 8; ++v)
        hout[(16 * mt + mbase + v) * NH + hcol] = hn[v];
    }
    __syncthreads();            // hlds[t] visible before next step's WMMAs
  }
}

// ---------------- decoder: [8192,256] x [256,10000] + bias ------------------
// A panel async-staged to LDS; all threads stage+barrier, inactive tail waves
// (nt >= 625) skip only the compute/store (wave-uniform, EXEC all-1 in WMMA).
__global__ __launch_bounds__(256)
void k_decoder(const __bf16* __restrict__ Ap, const __bf16* __restrict__ Bp,
               const float* __restrict__ decb, float* __restrict__ out) {
  __shared__ __align__(32) __bf16 alds[KC * 32 * 16];   // 8 KB
  int tid = threadIdx.x;
  int lane = tid & 31, w = tid >> 5;
  int mt = blockIdx.x;
  int nt = blockIdx.y * 8 + w;

  const void* apanel = (const void*)(Ap + (size_t)mt * KC * 32 * 16);
  unsigned lbase = (unsigned)(size_t)(&alds[0]);
  async_copy_b128(lbase + tid * 32,      apanel, (unsigned)(tid * 32));
  async_copy_b128(lbase + tid * 32 + 16, apanel, (unsigned)(tid * 32 + 16));
  wait_async0();
  __syncthreads();

  if (nt < NT_V) {
    int n = nt * 16 + (lane & 15);
    v8f acc = splat8(decb[n]);
    const v16bf* Bv = (const v16bf*)Bp;
    #pragma unroll
    for (int kc = 0; kc < KC; ++kc) {
      v16bf a = *(const v16bf*)&alds[(kc * 32 + lane) * 16];
      if (kc + 1 < KC)
        __builtin_prefetch(&Bv[((size_t)(kc + 1) * NT_V + nt) * 32 + lane], 0, 1);
      v16bf b = Bv[((size_t)kc * NT_V + nt) * 32 + lane];
      acc = wmma_bf16(a, b, acc);
    }
    int mb = 8 * (lane >> 4);
    #pragma unroll
    for (int v = 0; v < 8; ++v)
      out[(size_t)(mt * 16 + mb + v) * NVOC + n] = acc[v];
  }
}

// ---------------------------------------------------------------------------
extern "C" void kernel_launch(void* const* d_in, const int* in_sizes, int n_in,
                              void* d_out, int out_size, void* d_ws, size_t ws_size,
                              hipStream_t stream) {
  const int*   tokens = (const int*)d_in[0];
  const float* embW   = (const float*)d_in[1];
  const float* Wx0    = (const float*)d_in[2];
  const float* Wh0    = (const float*)d_in[3];
  const float* b0     = (const float*)d_in[4];
  const float* Wx1    = (const float*)d_in[5];
  const float* Wh1    = (const float*)d_in[6];
  const float* b1     = (const float*)d_in[7];
  const float* decW   = (const float*)d_in[8];
  const float* decb   = (const float*)d_in[9];
  float* out = (float*)d_out;

  // workspace carve-out (~53.4 MB total)
  char* ws = (char*)d_ws;
  size_t off = 0;
  auto carve = [&](size_t bytes) -> void* {
    void* p = ws + off;
    off = (off + bytes + 255) & ~(size_t)255;
    return p;
  };
  __bf16* xp    = (__bf16*)carve((size_t)SB * NIN * 2);
  float*  gxp   = (float*) carve((size_t)SB * NHG * 4);
  __bf16* h1p   = (__bf16*)carve((size_t)SB * NH * 2);
  __bf16* h2p   = (__bf16*)carve((size_t)SB * NH * 2);
  __bf16* Wx0p  = (__bf16*)carve((size_t)NH * NHG * 2);
  __bf16* Wh0p  = (__bf16*)carve((size_t)NH * NHG * 2);
  __bf16* Wx1p  = (__bf16*)carve((size_t)NH * NHG * 2);
  __bf16* Wh1p  = (__bf16*)carve((size_t)NH * NHG * 2);
  __bf16* decWp = (__bf16*)carve((size_t)NVOC * NH * 2);
  float*  b0p   = (float*) carve(NHG * 4);
  float*  b1p   = (float*) carve(NHG * 4);
  (void)ws_size; (void)in_sizes; (void)n_in; (void)out_size;

  // 1) pack weights / bias / embedding
  k_pack_gateW<<<64, 256, 0, stream>>>(Wx0, Wx0p);
  k_pack_gateW<<<64, 256, 0, stream>>>(Wh0, Wh0p);
  k_pack_gateW<<<64, 256, 0, stream>>>(Wx1, Wx1p);
  k_pack_gateW<<<64, 256, 0, stream>>>(Wh1, Wh1p);
  k_pack_decW<<<625, 256, 0, stream>>>(decW, decWp);
  k_pack_bias<<<4, 256, 0, stream>>>(b0, b0p);
  k_pack_bias<<<4, 256, 0, stream>>>(b1, b1p);
  k_embed_pack<<<512, 256, 0, stream>>>(tokens, embW, xp);

  // 2) layer 0
  k_gemm_gx<<<MT_TOT, 256, 0, stream>>>(xp, Wx0p, b0p, gxp);
  k_lstm<<<1, 1024, 0, stream>>>(gxp, Wh0p, h1p, out + OUT_H1);

  // 3) layer 1 (gx buffer reused)
  k_gemm_gx<<<MT_TOT, 256, 0, stream>>>(h1p, Wx1p, b1p, gxp);
  k_lstm<<<1, 1024, 0, stream>>>(gxp, Wh1p, h2p, out + OUT_H2);

  // 4) decoder
  dim3 dgrid(MT_TOT, (NT_V + 7) / 8);
  k_decoder<<<dgrid, 256, 0, stream>>>(h2p, decWp, decb, out + OUT_DEC);
}
